// ConvBnHTanhBin_57939108823748
// MI455X (gfx1250) — matrix operand order repaired and checked by reading, not compiled, not run on hardware
//
#include <hip/hip_runtime.h>
#include <hip/hip_bf16.h>

// ---------------------------------------------------------------------------
// Binary conv (sign(w)) + batchnorm(batch stats) + hardtanh + sign
// x: (32,256,56,56) f32, w: (256,256,3,3) f32, gamma/beta: (256,) f32
// Implicit GEMM, (r,s)-outer decomposition:
//   y[m][kout] = sum_{rs=0..8} sum_c x[n][c][h+r-1][w+s-1] * wb[kout][rs][c]
// M = 100352, N = 256, K per (rs) = 256 channels, bf16 WMMA f32-accum.
// ---------------------------------------------------------------------------

typedef __bf16 bf16_t;
typedef __attribute__((ext_vector_type(16))) __bf16 v16bf;
typedef __attribute__((ext_vector_type(8)))  __bf16 v8bf;
typedef __attribute__((ext_vector_type(8)))  float  v8f;

#define NN_   32
#define CC_   256
#define HH_   56
#define WW_   56
#define KOUT_ 256
#define HW_   (HH_ * WW_)          // 3136
#define MGEMM (NN_ * HW_)          // 100352
#define KGEMM (CC_ * 9)            // 2304

#define BM 128
#define BK 32
#define LDSA_STRIDE 40             // 32 + 8 pad (bf16), rows stay 16B aligned
#define LDSB_STRIDE 40

// ---------------------------------------------------------------------------
// Kernel 1: binarize + transpose weights OIHW -> bmat[kout][rs][c] bf16
// ---------------------------------------------------------------------------
__global__ __launch_bounds__(256) void k_binarize_w(const float* __restrict__ w,
                                                    bf16_t* __restrict__ bmat) {
    int idx  = blockIdx.x * 256 + threadIdx.x;   // 589824 = 2304 blocks exactly
    int kout = idx / KGEMM;
    int rem  = idx - kout * KGEMM;
    int rs   = rem / CC_;
    int c    = rem - rs * CC_;
    float v  = w[(kout * CC_ + c) * 9 + rs];
    float s  = (v > 0.0f) ? 1.0f : ((v < 0.0f) ? -1.0f : 0.0f);
    bmat[idx] = (bf16_t)s;                        // coalesced write
}

// ---------------------------------------------------------------------------
// Kernel 2: implicit-GEMM conv with bf16 WMMA.
// Grid: 784 blocks (MGEMM/BM), 256 threads = 8 wave32 waves.
// Block tile 128(M) x 256(N=all kout). Wave grid 2(M) x 4(N):
// each wave owns a 64x64 C tile = 4x4 WMMA tiles = 16 wmma / k-step.
// Outer loop: 9 filter taps (r,s); inner: 8 x 32-channel k-steps.
// y layout: [m][kout] f32.
// ---------------------------------------------------------------------------
__global__ __launch_bounds__(256)
void k_conv_wmma(const float* __restrict__ x, const bf16_t* __restrict__ bmat,
                 float* __restrict__ y) {
    __shared__ __align__(16) bf16_t ldsA[BM * LDSA_STRIDE];     // 10 KB
    __shared__ __align__(16) bf16_t ldsB[KOUT_ * LDSB_STRIDE];  // 20 KB

    const int tid   = threadIdx.x;
    const int lane  = tid & 31;
    const int wave  = tid >> 5;        // 0..7
    const int waveM = wave >> 2;       // 0..1 -> 64 rows
    const int waveN = wave & 3;        // 0..3 -> 64 cols
    const int lrow  = lane & 15;
    const int lhi   = lane >> 4;       // 0: K 0-7,16-23; 1: K 8-15,24-31

    const int gm0 = blockIdx.x * BM;

    // ---- A loader mapping: row i = lane + 32q (q=0..3), ch j = wave*4 + p
    const int jbase = wave * 4;
    int xbase[4], hq[4], wq[4];
#pragma unroll
    for (int q = 0; q < 4; ++q) {
        int gm = gm0 + lane + 32 * q;
        int n  = gm / HW_;
        int hw = gm - n * HW_;
        int h  = hw / WW_;
        hq[q]  = h;
        wq[q]  = hw - h * WW_;
        xbase[q] = n * (CC_ * HW_) + hw;   // n-base + h*56 + w
    }

    v8f acc[4][4] = {};

    for (int rs = 0; rs < 9; ++rs) {
        const int r   = rs / 3;            // uniform (scalar) per iteration
        const int s   = rs - r * 3;
        const int off = (r - 1) * WW_ + (s - 1);
        bool valid[4];
#pragma unroll
        for (int q = 0; q < 4; ++q)
            valid[q] = ((unsigned)(hq[q] + r - 1) < (unsigned)HH_) &&
                       ((unsigned)(wq[q] + s - 1) < (unsigned)WW_);

        const bf16_t* bsrc_row = bmat + ((size_t)tid * 9 + rs) * CC_;

        for (int c0 = 0; c0 < CC_; c0 += BK) {
            // ---- stage A tile: pure add + predicated load + cvt
#pragma unroll
            for (int q = 0; q < 4; ++q) {
                const float* xp = x + xbase[q] + (c0 + jbase) * HW_ + off;
                bf16_t* dst = &ldsA[(lane + 32 * q) * LDSA_STRIDE + jbase];
#pragma unroll
                for (int p = 0; p < 4; ++p) {
                    float v = valid[q] ? xp[p * HW_] : 0.0f;
                    dst[p] = (bf16_t)v;
                }
            }
            // ---- stage B tile: thread = kout row, 64B contiguous
            {
                const bf16_t* bsrc = bsrc_row + c0;
                bf16_t* dst = &ldsB[tid * LDSB_STRIDE];
#pragma unroll
                for (int p = 0; p < 4; ++p)
                    *(v8bf*)(dst + p * 8) = *(const v8bf*)(bsrc + p * 8);
                if (c0 + BK < CC_)
                    __builtin_prefetch(bsrc + BK, 0, 1);  // global_prefetch_b8
            }
            __syncthreads();

            // ---- fragments (ISA 16-bit A/B lane layout) + 16 WMMAs
            v16bf aF[4];
#pragma unroll
            for (int t = 0; t < 4; ++t) {
                int m = waveM * 64 + t * 16 + lrow;
                const bf16_t* p = &ldsA[m * LDSA_STRIDE + lhi * 8];
                v8bf lo = *(const v8bf*)(p);       // K = kh .. kh+7
                v8bf hi = *(const v8bf*)(p + 16);  // K = kh+16 .. kh+23
                aF[t] = __builtin_shufflevector(lo, hi, 0, 1, 2, 3, 4, 5, 6, 7,
                                                8, 9, 10, 11, 12, 13, 14, 15);
            }
            v16bf bF[4];
#pragma unroll
            for (int u = 0; u < 4; ++u) {
                int nn = waveN * 64 + u * 16 + lrow;
                const bf16_t* p = &ldsB[nn * LDSB_STRIDE + lhi * 8];
                v8bf lo = *(const v8bf*)(p);       // K = kh .. kh+7
                v8bf hi = *(const v8bf*)(p + 16);  // K = kh+16 .. kh+23
                bF[u] = __builtin_shufflevector(lo, hi, 0, 1, 2, 3, 4, 5, 6, 7,
                                                8, 9, 10, 11, 12, 13, 14, 15);
            }
#pragma unroll
            for (int t = 0; t < 4; ++t)
#pragma unroll
                for (int u = 0; u < 4; ++u)
                    acc[t][u] = __builtin_amdgcn_wmma_f32_16x16x32_bf16(
                        false, aF[t], false, bF[u], (short)0, acc[t][u],
                        false, false);

            __syncthreads();
        }
    }

    // ---- epilogue: C/D layout — lane%16 = col, VGPR v = row (+8 hi lanes)
#pragma unroll
    for (int t = 0; t < 4; ++t) {
        int gmRow = gm0 + waveM * 64 + t * 16 + (lhi ? 8 : 0);
#pragma unroll
        for (int u = 0; u < 4; ++u) {
            int gn = waveN * 64 + u * 16 + lrow;
#pragma unroll
            for (int v = 0; v < 8; ++v)
                y[(size_t)(gmRow + v) * KOUT_ + gn] = acc[t][u][v];
        }
    }
}

// ---------------------------------------------------------------------------
// Kernel 3: per-channel sum / sumsq over M. thread = channel, block = M chunk.
// stats: [0..255]=sum, [256..511]=sumsq (pre-zeroed by memset)
// ---------------------------------------------------------------------------
__global__ __launch_bounds__(256)
void k_bn_stats(const float* __restrict__ y, float* __restrict__ stats) {
    int k  = threadIdx.x;
    int m0 = blockIdx.x * 256;             // 392 blocks * 256 rows = MGEMM
    float s = 0.0f, sq = 0.0f;
    for (int m = m0; m < m0 + 256; ++m) {
        float v = y[(size_t)m * KOUT_ + k];
        s  += v;
        sq += v * v;
    }
    atomicAdd(&stats[k], s);
    atomicAdd(&stats[KOUT_ + k], sq);
}

// ---------------------------------------------------------------------------
// Kernel 4: finalize per-channel scale/shift. stats[512..767]=scale, [768..1023]=shift
// ---------------------------------------------------------------------------
__global__ __launch_bounds__(256)
void k_bn_finalize(float* __restrict__ stats, const float* __restrict__ gamma,
                   const float* __restrict__ beta) {
    int k = threadIdx.x;
    const float invM = 1.0f / (float)MGEMM;
    float mean = stats[k] * invM;
    float var  = stats[KOUT_ + k] * invM - mean * mean;
    float sc   = gamma[k] * rsqrtf(var + 1e-5f);
    stats[2 * KOUT_ + k] = sc;
    stats[3 * KOUT_ + k] = beta[k] - mean * sc;
}

// ---------------------------------------------------------------------------
// Kernel 5: BN + hardtanh + sign, with 32x32 LDS transpose [m][k] -> NCHW.
// Grid: (MGEMM/32 = 3136, KOUT_/32 = 8). 3136 % 32 == 0 so tiles never cross n.
// ---------------------------------------------------------------------------
__global__ __launch_bounds__(256)
void k_bn_apply(const float* __restrict__ y, const float* __restrict__ stats,
                float* __restrict__ out) {
    __shared__ float lds[32][33];
    const int m0 = blockIdx.x * 32;
    const int k0 = blockIdx.y * 32;
    const int row  = threadIdx.x >> 3;        // 0..31
    const int col0 = (threadIdx.x & 7) * 4;   // 0,4,...,28

    // coalesced read along k
    const float* yp = y + (size_t)(m0 + row) * KOUT_ + k0 + col0;
    float4 v4 = *(const float4*)yp;
    float vv[4] = {v4.x, v4.y, v4.z, v4.w};
#pragma unroll
    for (int j = 0; j < 4; ++j) {
        int k  = k0 + col0 + j;
        float t = vv[j] * stats[2 * KOUT_ + k] + stats[3 * KOUT_ + k];
        t = fminf(1.0f, fmaxf(-1.0f, t));
        t = (t > 0.0f) ? 1.0f : ((t < 0.0f) ? -1.0f : 0.0f);
        lds[row][col0 + j] = t;
    }
    __syncthreads();

    // coalesced write along hw (NCHW)
    int n   = m0 / HW_;
    int hw0 = m0 - n * HW_;
    float4 o;
    o.x = lds[col0 + 0][row];
    o.y = lds[col0 + 1][row];
    o.z = lds[col0 + 2][row];
    o.w = lds[col0 + 3][row];
    *(float4*)(out + (size_t)(n * KOUT_ + k0 + row) * HW_ + hw0 + col0) = o;
}

// ---------------------------------------------------------------------------
// Host-side launcher
// ---------------------------------------------------------------------------
extern "C" void kernel_launch(void* const* d_in, const int* in_sizes, int n_in,
                              void* d_out, int out_size, void* d_ws, size_t ws_size,
                              hipStream_t stream) {
    (void)in_sizes; (void)n_in; (void)out_size; (void)ws_size;

    const float* x     = (const float*)d_in[0];
    const float* w     = (const float*)d_in[1];
    const float* gamma = (const float*)d_in[2];
    const float* beta  = (const float*)d_in[3];
    float* out = (float*)d_out;

    // workspace carve-up
    const size_t ybytes = (size_t)MGEMM * KOUT_ * sizeof(float);      // ~98 MB
    const size_t bbytes = (size_t)KOUT_ * KGEMM * sizeof(bf16_t);     // ~1.2 MB
    float*  y     = (float*)d_ws;
    bf16_t* bmat  = (bf16_t*)((char*)d_ws + ybytes);
    float*  stats = (float*)((char*)d_ws + ybytes + bbytes);          // 1024 f32

    // 1) binarize + transpose weights -> bf16 [kout][rs][c]
    k_binarize_w<<<(KOUT_ * KGEMM) / 256, 256, 0, stream>>>(w, bmat);

    // 2) zero stats accumulators (graph-capturable)
    hipMemsetAsync(stats, 0, 2 * KOUT_ * sizeof(float), stream);

    // 3) implicit-GEMM conv via bf16 WMMA
    k_conv_wmma<<<MGEMM / BM, 256, 0, stream>>>(x, bmat, y);

    // 4) BN batch statistics + finalize
    k_bn_stats<<<MGEMM / 256, 256, 0, stream>>>(y, stats);
    k_bn_finalize<<<1, 256, 0, stream>>>(stats, gamma, beta);

    // 5) BN + hardtanh + sign, transpose to NCHW
    dim3 agrid(MGEMM / 32, KOUT_ / 32);
    k_bn_apply<<<agrid, 256, 0, stream>>>(y, stats, out);
}
